// CliqueEncoder_68049461838555
// MI455X (gfx1250) — matrix operand order repaired and checked by reading, not compiled, not run on hardware
//
#include <hip/hip_runtime.h>

// CDNA5 / gfx1250 wave32 WMMA implementation of the CliqueEncoder.
//
// out[row] = [ emb_table[t]  |  gaussian_basis(d) @ W[t] + b[t] ]
//
// Per wave: 16 rows = one WMMA M-tile, K = RBF = 32 (matches v_wmma_f32_16x16x32_f16).
// - gather + bias are expressed as one-hot(t) @ table WMMAs
// - per-type matmul = 4 WMMAs with type-masked basis A (packed 32-bit cndmask selects)
// B tiles staged once per block into LDS as f16 in the contiguous-k B layout used here (K = 16*h + j).
// Main loop covers only fully-in-bounds tiles (branch-free NT stores with const
// ioffsets); the <=15 tail rows are handled by a cold scalar f32 path on wave 0.

typedef __attribute__((ext_vector_type(16))) _Float16 v16h;
typedef __attribute__((ext_vector_type(8)))  float    v8f;
typedef __attribute__((ext_vector_type(8)))  unsigned v8u;
typedef __attribute__((ext_vector_type(2)))  int      v2i;

#define H_OUT     128
#define H2        64
#define RBF       32
#define NUM_TYPES 4
// 24 tiles: [0..3]=emb chunks, [4..7]=bias chunks, [8..23]=W[type][chunk]
#define NTILES    24

__global__ __launch_bounds__(256) void clique_encoder_wmma(
    const int*   __restrict__ attr,   // (N,2) int32: t, d
    const float* __restrict__ emb,    // (4,64)
    const float* __restrict__ W,      // (4,32,64)
    const float* __restrict__ bias,   // (4,64)
    float*       __restrict__ out,    // (N,128)
    int n)
{
  __shared__ __align__(32) _Float16 smem[NTILES * 512]; // 24 KB

  const int tid = threadIdx.x;

  // ---- one-time staging: f32 tables -> f16 WMMA-B-layout tiles in LDS ----
  // B order used here (32x16, contiguous k): lane L -> column N = L&15; halves j -> K = j + 16*(L>=16)
  for (int p = tid; p < NTILES * 32; p += 256) {
    const int tIdx = p >> 5;
    const int L    = p & 31;
    const int nCol = L & 15;
    const int kOff = (L & 16) ? 16 : 0;
    v16h vals;
#pragma unroll
    for (int j = 0; j < 16; ++j) {
      const int K = kOff + j;
      float v;
      if (tIdx < 4) {                       // emb chunk tile: rows K<4 = emb_table, rest 0
        v = (K < NUM_TYPES) ? emb[K * H2 + tIdx * 16 + nCol] : 0.0f;
      } else if (tIdx < 8) {                // bias chunk tile
        v = (K < NUM_TYPES) ? bias[K * H2 + (tIdx - 4) * 16 + nCol] : 0.0f;
      } else {                              // W[type] chunk tile
        const int ty = (tIdx - 8) >> 2;
        const int c  = (tIdx - 8) & 3;
        v = W[ty * (RBF * H2) + K * H2 + c * 16 + nCol];
      }
      vals[j] = (_Float16)v;
    }
    *(v16h*)&smem[(tIdx << 9) + (L << 4)] = vals;
  }
  __syncthreads();

  const int lane     = tid & 31;
  const int hi       = (lane >> 4) & 1;   // upper half-wave
  const int m        = lane & 15;         // row within 16-row tile
  const int waveId   = blockIdx.x * (blockDim.x >> 5) + (tid >> 5);
  const int numWaves = gridDim.x * (blockDim.x >> 5);
  const int numFull  = n >> 4;            // fully in-bounds 16-row tiles

  const float s    = 20.0f / 31.0f;       // RBF center spacing == std
  const float invS = 31.0f / 20.0f;

  // ================= hot loop: full tiles, branch-free =================
  for (int tile = waveId; tile < numFull; tile += numWaves) {
    const int rowBase = tile * 16;
    const v2i a2      = __builtin_nontemporal_load(&((const v2i*)attr)[rowBase + m]);
    const int   t     = a2.x;
    const float d     = (float)a2.y;

    // A order used here (16x32, contiguous k): lane -> M = lane&15; half j -> K = 16*hi + j
    v16h basisA;
#pragma unroll
    for (int j = 0; j < 16; ++j) {
      const int K = (hi << 4) + j;
      const float q = (d - (float)K * s) * invS;
      basisA[j] = (_Float16)__expf(-0.5f * q * q);
    }

    // one-hot(t): nonzero only in lanes 0-15 (K=j for j<4), at element j==t,
    // i.e. packed word t>>1, half t&1. Build directly as 32-bit words.
    const unsigned oneVal = 0x3c00u << ((t & 1) << 4);  // f16 1.0 in selected half
    v8u oneU = {};
    oneU[0] = ((hi == 0) && ((t >> 1) == 0)) ? oneVal : 0u;
    oneU[1] = ((hi == 0) && ((t >> 1) == 1)) ? oneVal : 0u;
    const v16h oneA = (v16h)oneU;

    // per-type masked basis: packed 32-bit selects (8 cndmask_b32 per type)
    const v8u basisU = (v8u)basisA;
    const v8u zeroU  = {};
    v16h maskA[NUM_TYPES];
#pragma unroll
    for (int i = 0; i < NUM_TYPES; ++i) {
      const v8u mu = (t == i) ? basisU : zeroU;
      maskA[i] = (v16h)mu;
    }

    // Per-tile store base: lane covers col=m of rows rowBase+8*hi .. +7
    float* const pB = out + (size_t)(rowBase + (hi << 3)) * H_OUT + m;

#pragma unroll
    for (int chunk = 0; chunk < 8; ++chunk) {
      v8f C = {};
      if (chunk < 4) {
        // shape half: C = onehot(t) @ emb_chunk  ==  emb_table[t] gather
        const v16h B = *(const v16h*)&smem[(chunk << 9) + (lane << 4)];
        C = __builtin_amdgcn_wmma_f32_16x16x32_f16(false, oneA, false, B,
                                                   (short)0, C, false, false);
      } else {
        const int c = chunk - 4;
        // bias seed: C = onehot(t) @ b_chunk
        const v16h Bb = *(const v16h*)&smem[((4 + c) << 9) + (lane << 4)];
        C = __builtin_amdgcn_wmma_f32_16x16x32_f16(false, oneA, false, Bb,
                                                   (short)0, C, false, false);
        // per-type masked basis @ W[type]_chunk, accumulated
#pragma unroll
        for (int i = 0; i < NUM_TYPES; ++i) {
          const v16h Bw = *(const v16h*)&smem[((8 + i * 4 + c) << 9) + (lane << 4)];
          C = __builtin_amdgcn_wmma_f32_16x16x32_f16(false, maskA[i], false, Bw,
                                                     (short)0, C, false, false);
        }
      }
      // C layout: VGPR r -> row (rowBase + 8*hi + r), col = chunk*16 + m
#pragma unroll
      for (int r = 0; r < 8; ++r)
        __builtin_nontemporal_store(C[r], pB + (chunk * 16 + r * H_OUT));
      // cap cross-chunk prefetch distance -> bound VGPR pressure
      __builtin_amdgcn_sched_barrier(0);
    }
  }

  // ================= cold tail: <=15 rows, scalar f32, wave 0 only =================
  if ((n & 15) && waveId == 0) {
    for (int row = numFull * 16; row < n; ++row) {
      const int   t = attr[2 * row];
      const float d = (float)attr[2 * row + 1];
      for (int col = lane; col < H_OUT; col += 32) {
        float v;
        if (col < H2) {
          v = emb[t * H2 + col];
        } else {
          float acc = bias[t * H2 + (col - H2)];
          for (int k = 0; k < RBF; ++k) {
            const float q = (d - (float)k * s) * invS;
            acc += __expf(-0.5f * q * q) * W[t * (RBF * H2) + k * H2 + (col - H2)];
          }
          v = acc;
        }
        out[(size_t)row * H_OUT + col] = v;
      }
    }
  }
}

extern "C" void kernel_launch(void* const* d_in, const int* in_sizes, int n_in,
                              void* d_out, int out_size, void* d_ws, size_t ws_size,
                              hipStream_t stream) {
  const int*   attr = (const int*)d_in[0];    // clique_attr (N,2) int32
  const float* emb  = (const float*)d_in[1];  // emb_table (4,64) f32
  const float* W    = (const float*)d_in[2];  // W (4,32,64) f32
  const float* bias = (const float*)d_in[3];  // b (4,64) f32
  float* out = (float*)d_out;

  const int n        = in_sizes[0] / 2;       // N rows
  const int numTiles = (n + 15) / 16;
  int blocks = (numTiles + 7) / 8;            // 8 waves/block, 1 tile/wave/iter
  if (blocks > 1024) blocks = 1024;
  if (blocks < 1) blocks = 1;

  clique_encoder_wmma<<<blocks, 256, 0, stream>>>(attr, emb, W, bias, out, n);
}
